// GMoE_network_noisy_1623497638191
// MI455X (gfx1250) — matrix operand order repaired and checked by reading, not compiled, not run on hardware
//
#include <hip/hip_runtime.h>
#include <math.h>

#define DF 64   // feature dim (D == H == 64)
#define NEXP 8
#define TOPK 4
#define LDS_WSTRIDE 72   // padded f16 row stride (16B-aligned chunks, breaks bank conflicts)

typedef __attribute__((ext_vector_type(16))) _Float16 v16h;
typedef __attribute__((ext_vector_type(8)))  _Float16 v8h;
typedef __attribute__((ext_vector_type(8)))  float    v8f;

// ---------------------------------------------------------------------------
// WMMA A fragment from LDS-staged f16 A tile (rows contiguous, stride 72):
//   lane<16 -> M=rowLocal+lane; per lane two contiguous 8xf16 (16B) chunks at
//   K = ks*32 + hi*8 and K = ks*32 + 16 + hi*8  ->  2x ds_load_b128
// ---------------------------------------------------------------------------
__device__ __forceinline__ v16h load_a_frag_lds(const _Float16* __restrict__ at,
                                                int rowLocal, int ks, int lane) {
  int m = rowLocal + (lane & 15);
  const int hi = (lane >> 4) & 1;
  const _Float16* p = at + (size_t)m * LDS_WSTRIDE + ks * 32 + hi * 8;
  const v8h* pv0 = (const v8h*)p;         // 16B aligned (row stride 144B, off {0,16,64,80}B)
  const v8h* pv1 = (const v8h*)(p + 16);
  v8h lo = pv0[0];
  v8h hs = pv1[0];
  v16h f;
#pragma unroll
  for (int i = 0; i < 8; ++i) { f[i] = lo[i]; f[8 + i] = hs[i]; }
  return f;
}

// B fragment from LDS-staged transposed f16 W (WT[n][k], stride LDS_WSTRIDE):
//   lane<16 -> N=lane, two 8xf16 chunks at K = ks*32 + hi*8 and K = ks*32 + 16 + hi*8 -> 2x ds_load_b128
__device__ __forceinline__ v16h load_b_frag_lds(const _Float16* __restrict__ wt,
                                                int colBase, int ks, int lane) {
  int n = colBase + (lane & 15);
  const int hi = (lane >> 4) & 1;
  const _Float16* p = wt + (size_t)n * LDS_WSTRIDE + ks * 32 + hi * 8;
  const v8h* pv = (const v8h*)p;
  v8h lo = pv[0];
  v8h hs = pv[2];
  v16h f;
#pragma unroll
  for (int i = 0; i < 8; ++i) { f[i] = lo[i]; f[8 + i] = hs[i]; }
  return f;
}

// ---------------------------------------------------------------------------
// GEMM: out(N x 64) = sum_i A_i(N x 64) @ W_i(64 x 64) [+ bias] [relu]
// 256 threads = 8 waves; block covers 32 rows (2 row tiles x 4 col tiles).
// Both the 32x64 A tile and 64x64 W (transposed) are staged into LDS as f16
// once per block (coalesced global reads, one convert per element); each wave
// then runs 2 k-steps of v_wmma_f32_16x16x32_f16 per input with pure
// ds_load_b128 operand fetches.
// ---------------------------------------------------------------------------
template <int NIN, bool RELU>
__global__ void __launch_bounds__(256)
gemm64_wmma(const float* __restrict__ A0, const float* __restrict__ W0,
            const float* __restrict__ A1, const float* __restrict__ W1,
            const float* __restrict__ A2, const float* __restrict__ W2,
            const float* __restrict__ bias, float* __restrict__ out, int n) {
  __shared__ _Float16 smemW[NIN * DF * LDS_WSTRIDE];
  __shared__ _Float16 smemA[NIN * 32 * LDS_WSTRIDE];

  const int blockRow = blockIdx.x * 32;

  // cooperative staging
#pragma unroll
  for (int w = 0; w < NIN; ++w) {
    const float* Wp = (w == 0) ? W0 : (w == 1) ? W1 : W2;
    const float* Ap = (w == 0) ? A0 : (w == 1) ? A1 : A2;
    _Float16* dstw = smemW + w * (DF * LDS_WSTRIDE);
    _Float16* dsta = smemA + w * (32 * LDS_WSTRIDE);
    // WT[n][k] = (f16) W[k][n]   (consecutive threads -> consecutive n)
    for (int kn = threadIdx.x; kn < DF * DF; kn += 256) {
      int k = kn >> 6, c = kn & 63;
      dstw[c * LDS_WSTRIDE + k] = (_Float16)Wp[kn];
    }
    // A tile rows (clamped at tail)
    for (int idx = threadIdx.x; idx < 32 * DF; idx += 256) {
      int r = idx >> 6, c = idx & 63;
      int gr = blockRow + r;
      if (gr >= n) gr = n - 1;
      dsta[r * LDS_WSTRIDE + c] = (_Float16)Ap[(size_t)gr * DF + c];
    }
  }
  __syncthreads();

  const int lane = threadIdx.x & 31;
  const int wave = threadIdx.x >> 5;
  const int rowLocal = (wave >> 2) * 16;
  const int rowBase = blockRow + rowLocal;
  const int colBase = (wave & 3) * 16;

  v8f acc;
  float bv = bias ? bias[colBase + (lane & 15)] : 0.0f;
#pragma unroll
  for (int i = 0; i < 8; ++i) acc[i] = bv;

#pragma unroll
  for (int ks = 0; ks < 2; ++ks) {
    v16h a = load_a_frag_lds(smemA, rowLocal, ks, lane);
    v16h b = load_b_frag_lds(smemW, colBase, ks, lane);
    acc = __builtin_amdgcn_wmma_f32_16x16x32_f16(false, a, false, b, (short)0, acc,
                                                 false, false);
  }
  if constexpr (NIN >= 2) {
#pragma unroll
    for (int ks = 0; ks < 2; ++ks) {
      v16h a = load_a_frag_lds(smemA + 32 * LDS_WSTRIDE, rowLocal, ks, lane);
      v16h b = load_b_frag_lds(smemW + DF * LDS_WSTRIDE, colBase, ks, lane);
      acc = __builtin_amdgcn_wmma_f32_16x16x32_f16(false, a, false, b, (short)0, acc,
                                                   false, false);
    }
  }
  if constexpr (NIN >= 3) {
#pragma unroll
    for (int ks = 0; ks < 2; ++ks) {
      v16h a = load_a_frag_lds(smemA + 2 * 32 * LDS_WSTRIDE, rowLocal, ks, lane);
      v16h b = load_b_frag_lds(smemW + 2 * DF * LDS_WSTRIDE, colBase, ks, lane);
      acc = __builtin_amdgcn_wmma_f32_16x16x32_f16(false, a, false, b, (short)0, acc,
                                                   false, false);
    }
  }

  // epilogue: wave-uniform full-tile fast path (single scalar branch)
  const int ncol = colBase + (lane & 15);
  const int moff = (lane >> 4) ? 8 : 0;
  if (rowBase + 16 <= n) {
    float* obase = out + (size_t)rowBase * DF + ncol;
#pragma unroll
    for (int v = 0; v < 8; ++v) {
      float r = acc[v];
      if (RELU) r = fmaxf(r, 0.0f);
      obase[(size_t)(v + moff) * DF] = r;
    }
  } else {
#pragma unroll
    for (int v = 0; v < 8; ++v) {
      int m = rowBase + v + moff;
      float r = acc[v];
      if (RELU) r = fmaxf(r, 0.0f);
      if (m < n) out[(size_t)m * DF + ncol] = r;
    }
  }
}

// ---------------------------------------------------------------------------
// Elementwise / fill kernels
// ---------------------------------------------------------------------------
__global__ void fill_kernel(float* __restrict__ p, float v, int n) {
  int i = blockIdx.x * blockDim.x + threadIdx.x;
  if (i < n) p[i] = v;
}
__global__ void sub_kernel(const float* __restrict__ a, const float* __restrict__ b,
                           float* __restrict__ c, int n) {
  int i = blockIdx.x * blockDim.x + threadIdx.x;
  if (i < n) c[i] = a[i] - b[i];
}
__global__ void relu_kernel(float* __restrict__ a, int n) {
  int i = blockIdx.x * blockDim.x + threadIdx.x;
  if (i < n) a[i] = fmaxf(a[i], 0.0f);
}
__global__ void gin_self_kernel(float* __restrict__ out, const float* __restrict__ x,
                                const float* __restrict__ eps, int n) {
  int i = blockIdx.x * blockDim.x + threadIdx.x;
  if (i < n) out[i] += (1.0f + eps[0]) * x[i];
}
__global__ void meanscale_kernel(float* __restrict__ out, const float* __restrict__ inv,
                                 int n) {
  int i = blockIdx.x * blockDim.x + threadIdx.x;
  if (i < n) out[i] *= inv[i >> 6];
}
__global__ void accum_kernel(const float* __restrict__ eo, const float* __restrict__ gates,
                             float* __restrict__ y, int e, int n) {
  int i = blockIdx.x * blockDim.x + threadIdx.x;
  if (i < n) y[i] += gates[(size_t)(i >> 6) * NEXP + e] * eo[i] * 0.125f;
}

// ---------------------------------------------------------------------------
// Degree / normalization
// ---------------------------------------------------------------------------
__global__ void deg_kernel(const int* __restrict__ dst, float* __restrict__ deg, int E) {
  int i = blockIdx.x * blockDim.x + threadIdx.x;
  if (i < E) atomicAdd(&deg[dst[i]], 1.0f);
}
__global__ void deg_fin_kernel(const float* __restrict__ deg, float* __restrict__ rs,
                               float* __restrict__ inv, int n) {
  int i = blockIdx.x * blockDim.x + threadIdx.x;
  if (i < n) {
    float d = fmaxf(deg[i], 1.0f);
    rs[i] = rsqrtf(d);
    inv[i] = 1.0f / d;
  }
}

// ---------------------------------------------------------------------------
// Edge scatter kernels (warp-per-edge, 32 lanes x 2 features)
// ---------------------------------------------------------------------------
__global__ void scatter_gcn_kernel(const float* __restrict__ x, const int* __restrict__ src,
                                   const int* __restrict__ dst, const float* __restrict__ rs,
                                   float* __restrict__ out, int E) {
  int gid = blockIdx.x * blockDim.x + threadIdx.x;
  int e = gid >> 5, lane = gid & 31;
  if (e >= E) return;
  int s = src[e], d = dst[e];
  float w = rs[s] * rs[d];
  atomicAdd(&out[(size_t)d * DF + lane],      x[(size_t)s * DF + lane] * w);
  atomicAdd(&out[(size_t)d * DF + lane + 32], x[(size_t)s * DF + lane + 32] * w);
}
__global__ void scatter_sum_kernel(const float* __restrict__ x, const int* __restrict__ src,
                                   const int* __restrict__ dst, float* __restrict__ out,
                                   int E) {
  int gid = blockIdx.x * blockDim.x + threadIdx.x;
  int e = gid >> 5, lane = gid & 31;
  if (e >= E) return;
  int s = src[e], d = dst[e];
  atomicAdd(&out[(size_t)d * DF + lane],      x[(size_t)s * DF + lane]);
  atomicAdd(&out[(size_t)d * DF + lane + 32], x[(size_t)s * DF + lane + 32]);
}

// ---------------------------------------------------------------------------
// GAT attention
// ---------------------------------------------------------------------------
__device__ __forceinline__ void atomicMaxFloat(float* addr, float v) {
  if (v >= 0.0f) atomicMax((int*)addr, __float_as_int(v));
  else           atomicMin((unsigned int*)addr, __float_as_uint(v));
}
__global__ void gat_scalar_kernel(const float* __restrict__ h, const float* __restrict__ as_,
                                  const float* __restrict__ ad_, float* __restrict__ es,
                                  float* __restrict__ ed, int n) {
  int i = blockIdx.x * blockDim.x + threadIdx.x;
  if (i >= n) return;
  float s = 0.0f, d = 0.0f;
  const float* row = h + (size_t)i * DF;
  for (int k = 0; k < DF; ++k) { float v = row[k]; s += v * as_[k]; d += v * ad_[k]; }
  es[i] = s; ed[i] = d;
}
__global__ void gat_edge1_kernel(const float* __restrict__ es, const float* __restrict__ ed,
                                 const int* __restrict__ src, const int* __restrict__ dst,
                                 float* __restrict__ ebuf, float* __restrict__ mbuf, int E) {
  int i = blockIdx.x * blockDim.x + threadIdx.x;
  if (i >= E) return;
  float e = es[src[i]] + ed[dst[i]];
  e = (e > 0.0f) ? e : 0.2f * e;             // leaky relu 0.2
  ebuf[i] = e;
  atomicMaxFloat(&mbuf[dst[i]], e);
}
__global__ void gat_edge2_kernel(float* __restrict__ ebuf, const float* __restrict__ mbuf,
                                 const int* __restrict__ dst, float* __restrict__ sbuf,
                                 int E) {
  int i = blockIdx.x * blockDim.x + threadIdx.x;
  if (i >= E) return;
  float e = expf(ebuf[i] - mbuf[dst[i]]);
  ebuf[i] = e;
  atomicAdd(&sbuf[dst[i]], e);
}
__global__ void gat_edge3_kernel(const float* __restrict__ ebuf, const float* __restrict__ sbuf,
                                 const float* __restrict__ h, const int* __restrict__ src,
                                 const int* __restrict__ dst, float* __restrict__ out, int E) {
  int gid = blockIdx.x * blockDim.x + threadIdx.x;
  int e = gid >> 5, lane = gid & 31;
  if (e >= E) return;
  int s = src[e], d = dst[e];
  float w = ebuf[e] / (sbuf[d] + 1e-9f);
  atomicAdd(&out[(size_t)d * DF + lane],      w * h[(size_t)s * DF + lane]);
  atomicAdd(&out[(size_t)d * DF + lane + 32], w * h[(size_t)s * DF + lane + 32]);
}

// ---------------------------------------------------------------------------
// Gating: clean/noisy, top-5, softmax top-4, load (erf CDF); per-block
// reductions of gates.sum(0) and load.sum(0) into gsum[16].
// ---------------------------------------------------------------------------
__global__ void gate_kernel(const float* __restrict__ x, const float* __restrict__ wg,
                            const float* __restrict__ wn, const float* __restrict__ noise,
                            float* __restrict__ gates, float* __restrict__ gsum, int n) {
  __shared__ float sh[16];
  if (threadIdx.x < 16) sh[threadIdx.x] = 0.0f;
  __syncthreads();
  int node = blockIdx.x * blockDim.x + threadIdx.x;
  if (node < n) {
    float clean[NEXP], nz[NEXP], noisy[NEXP], std_[NEXP];
#pragma unroll
    for (int e = 0; e < NEXP; ++e) { clean[e] = 0.0f; nz[e] = 0.0f; }
    const float* row = x + (size_t)node * DF;
    for (int k = 0; k < DF; ++k) {
      float xv = row[k];
#pragma unroll
      for (int e = 0; e < NEXP; ++e) {
        clean[e] += xv * wg[k * NEXP + e];
        nz[e]    += xv * wn[k * NEXP + e];
      }
    }
#pragma unroll
    for (int e = 0; e < NEXP; ++e) {
      float sp = logf(1.0f + expf(nz[e]));   // softplus
      std_[e] = sp + 0.01f;
      noisy[e] = clean[e] + noise[(size_t)node * NEXP + e] * std_[e];
    }
    // top-(K+1)=5 by selection (ties -> lowest index, like lax.top_k)
    float tmp[NEXP];
#pragma unroll
    for (int e = 0; e < NEXP; ++e) tmp[e] = noisy[e];
    float tv[TOPK + 1]; int ti[TOPK + 1];
    for (int j = 0; j < TOPK + 1; ++j) {
      int bi = 0; float bvv = tmp[0];
      for (int e = 1; e < NEXP; ++e) if (tmp[e] > bvv) { bvv = tmp[e]; bi = e; }
      tv[j] = bvv; ti[j] = bi; tmp[bi] = -INFINITY;
    }
    // softmax over top-4
    float mx = tv[0], se = 0.0f, ex[TOPK];
    for (int j = 0; j < TOPK; ++j) { ex[j] = expf(tv[j] - mx); se += ex[j]; }
    float g[NEXP];
#pragma unroll
    for (int e = 0; e < NEXP; ++e) g[e] = 0.0f;
    for (int j = 0; j < TOPK; ++j) g[ti[j]] = ex[j] / se;
#pragma unroll
    for (int e = 0; e < NEXP; ++e) gates[(size_t)node * NEXP + e] = g[e];
    // load (normal CDF with threshold in/out)
    float thrin = tv[TOPK], throut = tv[TOPK - 1];
#pragma unroll
    for (int e = 0; e < NEXP; ++e) {
      float z = (noisy[e] > thrin) ? (clean[e] - thrin) / std_[e]
                                   : (clean[e] - throut) / std_[e];
      float lv = 0.5f * (1.0f + erff(z * 0.70710678118654752f));
      atomicAdd(&sh[e], g[e]);
      atomicAdd(&sh[NEXP + e], lv);
    }
  }
  __syncthreads();
  if (threadIdx.x < 16) atomicAdd(&gsum[threadIdx.x], sh[threadIdx.x]);
}

__global__ void loss_kernel(const float* __restrict__ gsum, float* __restrict__ out) {
  float total = 0.0f;
  for (int part = 0; part < 2; ++part) {
    const float* v = gsum + part * NEXP;
    float mean = 0.0f;
    for (int e = 0; e < NEXP; ++e) mean += v[e];
    mean /= (float)NEXP;
    float var = 0.0f;
    for (int e = 0; e < NEXP; ++e) { float d = v[e] - mean; var += d * d; }
    var /= (float)(NEXP - 1);                 // ddof=1
    total += var / (mean * mean + 1e-10f);
  }
  out[0] = 0.01f * total;
}

// ---------------------------------------------------------------------------
// Host-side orchestration
// ---------------------------------------------------------------------------
static inline int cdiv(int a, int b) { return (a + b - 1) / b; }

extern "C" void kernel_launch(void* const* d_in, const int* in_sizes, int n_in,
                              void* d_out, int out_size, void* d_ws, size_t ws_size,
                              hipStream_t stream) {
  const float* x     = (const float*)d_in[0];
  const int*   ei    = (const int*)d_in[1];
  const float* noise = (const float*)d_in[2];
  const float* wg    = (const float*)d_in[3];
  const float* wn    = (const float*)d_in[4];
  int ii = 5;
  const float* gcnW1 = (const float*)d_in[ii++]; const float* gcnB1 = (const float*)d_in[ii++];
  const float* gcnW2 = (const float*)d_in[ii++]; const float* gcnB2 = (const float*)d_in[ii++];
  const float* gatW1 = (const float*)d_in[ii++]; const float* gatAs1 = (const float*)d_in[ii++];
  const float* gatAd1 = (const float*)d_in[ii++]; const float* gatW2 = (const float*)d_in[ii++];
  const float* gatAs2 = (const float*)d_in[ii++]; const float* gatAd2 = (const float*)d_in[ii++];
  const float* sWs1 = (const float*)d_in[ii++]; const float* sWn1 = (const float*)d_in[ii++];
  const float* sB1  = (const float*)d_in[ii++]; const float* sWs2 = (const float*)d_in[ii++];
  const float* sWn2 = (const float*)d_in[ii++]; const float* sB2  = (const float*)d_in[ii++];
  const float* sgcW = (const float*)d_in[ii++]; const float* sgcB = (const float*)d_in[ii++];
  const float* jkW1 = (const float*)d_in[ii++]; const float* jkW2 = (const float*)d_in[ii++];
  const float* jkWjk = (const float*)d_in[ii++];
  const float* chW10 = (const float*)d_in[ii++]; const float* chW11 = (const float*)d_in[ii++];
  const float* chW20 = (const float*)d_in[ii++]; const float* chW21 = (const float*)d_in[ii++];
  const float* ginE1 = (const float*)d_in[ii++]; const float* ginW11 = (const float*)d_in[ii++];
  const float* ginW12 = (const float*)d_in[ii++]; const float* ginE2 = (const float*)d_in[ii++];
  const float* ginW21 = (const float*)d_in[ii++]; const float* ginW22 = (const float*)d_in[ii++];
  const float* mhW0 = (const float*)d_in[ii++]; const float* mhW1 = (const float*)d_in[ii++];
  const float* mhW2 = (const float*)d_in[ii++]; const float* mhWout = (const float*)d_in[ii++];

  const int N = in_sizes[0] / DF;
  const int E = in_sizes[1] / 2;
  const int* src = ei;
  const int* dst = ei + E;

  // workspace carve-up
  float* ws = (float*)d_ws;
  size_t off = 0;
  auto alloc = [&](size_t cnt) { float* p = ws + off; off += cnt; return p; };
  float* rs   = alloc(N);
  float* inv  = alloc(N);
  float* es   = alloc(N);
  float* ed   = alloc(N);
  float* mbuf = alloc(N);
  float* sbuf = alloc(N);     // also used as degree temp
  float* gates = alloc((size_t)N * NEXP);
  float* gsum = alloc(16);
  float* ebuf = alloc(E);
  float* A1 = alloc((size_t)N * DF);
  float* A2 = alloc((size_t)N * DF);
  float* t0 = alloc((size_t)N * DF);
  float* t1 = alloc((size_t)N * DF);
  float* t2 = alloc((size_t)N * DF);
  float* Eo = alloc((size_t)N * DF);
  (void)ws_size; (void)n_in; (void)out_size;

  float* y = (float*)d_out;            // N*DF outputs, then 1 loss scalar
  const int NF = N * DF;
  const int TB = 256;
  const int gridNF = cdiv(NF, TB);
  const int gridN  = cdiv(N, TB);
  const int gridE  = cdiv(E, TB);
  const int gridEw = cdiv(E, TB / 32);     // warp-per-edge
  const int gridG  = cdiv(N, 32);          // GEMM: 32 rows per block

#define FILL(p, v, c) fill_kernel<<<cdiv((c), TB), TB, 0, stream>>>((p), (v), (c))
#define GEMM1(A, W, B, O, R) gemm64_wmma<1, R><<<gridG, TB, 0, stream>>>((A), (W), nullptr, nullptr, nullptr, nullptr, (B), (O), N)
#define GEMM2(Aa, Wa, Ab, Wb, B, O, R) gemm64_wmma<2, R><<<gridG, TB, 0, stream>>>((Aa), (Wa), (Ab), (Wb), nullptr, nullptr, (B), (O), N)
#define GEMM3(Aa, Wa, Ab, Wb, Ac, Wc, B, O, R) gemm64_wmma<3, R><<<gridG, TB, 0, stream>>>((Aa), (Wa), (Ab), (Wb), (Ac), (Wc), (B), (O), N)

  // ---- degrees / norm ----
  FILL(sbuf, 0.0f, N);
  deg_kernel<<<gridE, TB, 0, stream>>>(dst, sbuf, E);
  deg_fin_kernel<<<gridN, TB, 0, stream>>>(sbuf, rs, inv, N);

  // ---- gating + y init ----
  FILL(gsum, 0.0f, 16);
  gate_kernel<<<cdiv(N, 128), 128, 0, stream>>>(x, wg, wn, noise, gates, gsum, N);
  FILL(y, 0.0f, NF);

  // ---- shared propagation chains: a1 = prop(x), a2 = prop(a1) ----
  FILL(A1, 0.0f, NF);
  scatter_gcn_kernel<<<gridEw, TB, 0, stream>>>(x, src, dst, rs, A1, E);
  FILL(A2, 0.0f, NF);
  scatter_gcn_kernel<<<gridEw, TB, 0, stream>>>(A1, src, dst, rs, A2, E);

  // ---- expert 0: GCN ----
  GEMM1(A1, gcnW1, gcnB1, t0, true);
  FILL(t1, 0.0f, NF);
  scatter_gcn_kernel<<<gridEw, TB, 0, stream>>>(t0, src, dst, rs, t1, E);
  GEMM1(t1, gcnW2, gcnB2, Eo, false);
  accum_kernel<<<gridNF, TB, 0, stream>>>(Eo, gates, y, 0, NF);

  // ---- expert 3: SGC ----
  GEMM1(A2, sgcW, sgcB, Eo, false);
  accum_kernel<<<gridNF, TB, 0, stream>>>(Eo, gates, y, 3, NF);

  // ---- expert 4: JK ----
  GEMM1(A1, jkW1, nullptr, t0, true);                 // h1
  FILL(t1, 0.0f, NF);
  scatter_gcn_kernel<<<gridEw, TB, 0, stream>>>(t0, src, dst, rs, t1, E);
  GEMM1(t1, jkW2, nullptr, t2, true);                 // h2
  GEMM2(t0, jkWjk, t2, jkWjk + 64 * 64, nullptr, Eo, false);
  accum_kernel<<<gridNF, TB, 0, stream>>>(Eo, gates, y, 4, NF);

  // ---- expert 7: MixHop ----
  GEMM1(x, mhW0, nullptr, t0, true);
  GEMM1(A1, mhW1, nullptr, t1, true);
  GEMM1(A2, mhW2, nullptr, t2, true);
  GEMM3(t0, mhWout, t1, mhWout + 64 * 64, t2, mhWout + 128 * 64, nullptr, Eo, false);
  accum_kernel<<<gridNF, TB, 0, stream>>>(Eo, gates, y, 7, NF);

  // ---- expert 5: Cheb ----
  sub_kernel<<<gridNF, TB, 0, stream>>>(x, A1, t0, NF);          // t1 = x - prop(x)
  GEMM2(x, chW10, t0, chW11, nullptr, t1, true);                 // h
  FILL(t2, 0.0f, NF);
  scatter_gcn_kernel<<<gridEw, TB, 0, stream>>>(t1, src, dst, rs, t2, E);
  sub_kernel<<<gridNF, TB, 0, stream>>>(t1, t2, t2, NF);         // t1b = h - prop(h)
  GEMM2(t1, chW20, t2, chW21, nullptr, Eo, false);
  accum_kernel<<<gridNF, TB, 0, stream>>>(Eo, gates, y, 5, NF);

  // ---- expert 6: GIN ----
  FILL(t0, 0.0f, NF);
  scatter_sum_kernel<<<gridEw, TB, 0, stream>>>(x, src, dst, t0, E);
  gin_self_kernel<<<gridNF, TB, 0, stream>>>(t0, x, ginE1, NF);
  GEMM1(t0, ginW11, nullptr, t1, true);
  GEMM1(t1, ginW12, nullptr, t2, true);                          // h
  FILL(t0, 0.0f, NF);
  scatter_sum_kernel<<<gridEw, TB, 0, stream>>>(t2, src, dst, t0, E);
  gin_self_kernel<<<gridNF, TB, 0, stream>>>(t0, t2, ginE2, NF);
  GEMM1(t0, ginW21, nullptr, t1, true);
  GEMM1(t1, ginW22, nullptr, Eo, false);
  accum_kernel<<<gridNF, TB, 0, stream>>>(Eo, gates, y, 6, NF);

  // ---- expert 2: SAGE ----
  FILL(t0, 0.0f, NF);
  scatter_sum_kernel<<<gridEw, TB, 0, stream>>>(x, src, dst, t0, E);
  meanscale_kernel<<<gridNF, TB, 0, stream>>>(t0, inv, NF);       // mean_agg(x)
  GEMM2(x, sWs1, t0, sWn1, sB1, t1, true);                        // h
  FILL(t2, 0.0f, NF);
  scatter_sum_kernel<<<gridEw, TB, 0, stream>>>(t1, src, dst, t2, E);
  meanscale_kernel<<<gridNF, TB, 0, stream>>>(t2, inv, NF);       // mean_agg(h)
  GEMM2(t1, sWs2, t2, sWn2, sB2, Eo, false);
  accum_kernel<<<gridNF, TB, 0, stream>>>(Eo, gates, y, 2, NF);

  // ---- expert 1: GAT ----
  // layer 1
  GEMM1(x, gatW1, nullptr, t0, false);                            // h = x@W1
  gat_scalar_kernel<<<gridN, TB, 0, stream>>>(t0, gatAs1, gatAd1, es, ed, N);
  FILL(mbuf, -INFINITY, N);
  FILL(sbuf, 0.0f, N);
  FILL(t1, 0.0f, NF);
  gat_edge1_kernel<<<gridE, TB, 0, stream>>>(es, ed, src, dst, ebuf, mbuf, E);
  gat_edge2_kernel<<<gridE, TB, 0, stream>>>(ebuf, mbuf, dst, sbuf, E);
  gat_edge3_kernel<<<gridEw, TB, 0, stream>>>(ebuf, sbuf, t0, src, dst, t1, E);
  relu_kernel<<<gridNF, TB, 0, stream>>>(t1, NF);
  // layer 2
  GEMM1(t1, gatW2, nullptr, t0, false);
  gat_scalar_kernel<<<gridN, TB, 0, stream>>>(t0, gatAs2, gatAd2, es, ed, N);
  FILL(mbuf, -INFINITY, N);
  FILL(sbuf, 0.0f, N);
  FILL(Eo, 0.0f, NF);
  gat_edge1_kernel<<<gridE, TB, 0, stream>>>(es, ed, src, dst, ebuf, mbuf, E);
  gat_edge2_kernel<<<gridE, TB, 0, stream>>>(ebuf, mbuf, dst, sbuf, E);
  gat_edge3_kernel<<<gridEw, TB, 0, stream>>>(ebuf, sbuf, t0, src, dst, Eo, E);
  accum_kernel<<<gridNF, TB, 0, stream>>>(Eo, gates, y, 1, NF);

  // ---- loss ----
  loss_kernel<<<1, 1, 0, stream>>>(gsum, y + (size_t)NF);

#undef FILL
#undef GEMM1
#undef GEMM2
#undef GEMM3
}
